// MLP_CLIP_80367428042763
// MI455X (gfx1250) — matrix operand order repaired and checked by reading, not compiled, not run on hardware
//
#include <hip/hip_runtime.h>
#include <hip/hip_bf16.h>

// ---------------------------------------------------------------------------
// MLP_CLIP fused pipeline for MI455X (gfx1250).
// bf16 WMMA + TDM (tensor_load_to_lds, TENSORcnt) for B tiles +
// GLOBAL_LOAD_ASYNC_TO_LDS (ASYNCcnt) for A tiles, double-buffered.
// B=16384, D=512, NTXT=4096, H=1024.
// ---------------------------------------------------------------------------

typedef __attribute__((ext_vector_type(16))) __bf16 v16bf;
typedef __attribute__((ext_vector_type(8)))  float  v8f;
typedef __attribute__((ext_vector_type(4)))  unsigned v4u;
typedef __attribute__((ext_vector_type(8)))  int      v8i;
typedef __attribute__((ext_vector_type(4)))  int      v4i;

#define B_SZ  16384
#define D_SZ  512
#define N_TXT 4096
#define H_SZ  1024

#define KT    64            // K-tile per staged buffer
#define LDSTR 72            // padded LDS row stride (bf16), 144B rows
#define A_ELE (64  * LDSTR) // A tile elems per buffer
#define B_ELE (128 * LDSTR) // B tile elems per buffer

#if __has_builtin(__builtin_amdgcn_tensor_load_to_lds)
#define HAVE_TDM 1
#else
#define HAVE_TDM 0
#endif

union V16BF { uint4 q[2]; v16bf v; };

__device__ __forceinline__ unsigned short f32_to_bf16(float f) {
  unsigned u = __float_as_uint(f);
  unsigned r = (u + 0x7FFFu + ((u >> 16) & 1u)) >> 16;  // round-to-nearest-even
  return (unsigned short)r;
}

// ---- CDNA5 async copy global->LDS (ASYNCcnt) ------------------------------
__device__ __forceinline__ void async_ld_b128(unsigned lds_off, const unsigned short* g) {
  asm volatile("global_load_async_to_lds_b128 %0, %1, off"
               :: "v"(lds_off), "v"(g) : "memory");
}

template <int N>
__device__ __forceinline__ void wait_asynccnt() {
#if __has_builtin(__builtin_amdgcn_s_wait_asynccnt)
  __builtin_amdgcn_s_wait_asynccnt(N);
#else
  asm volatile("s_wait_asynccnt %0" :: "i"(N));
#endif
}

template <int N>
__device__ __forceinline__ void wait_tensorcnt() {
#if __has_builtin(__builtin_amdgcn_s_wait_tensorcnt)
  __builtin_amdgcn_s_wait_tensorcnt((short)N);
#else
  asm volatile("s_wait_tensorcnt %0" :: "i"(N));
#endif
}

__device__ __forceinline__ unsigned lds_off_of(const void* p) {
  // LDS aperture: low 32 bits of the generic address are the LDS byte offset.
  return (unsigned)(size_t)p;
}

// ---- TDM: 2D tile load (bf16 elements) with LDS row padding ---------------
// Loads tile_d1 rows x tile_d0 elems from a row-major tensor (row stride
// stride_d0 elems) into LDS, inserting 16B of pad after every 128B stored
// (pad_interval=32 DW, pad_amount=4 DW) -> reproduces LDSTR=72 layout.
#if HAVE_TDM
__device__ __forceinline__ void tdm_load_2d(unsigned lds_addr, const void* gaddr,
                                            unsigned tensor_d0, unsigned tensor_d1,
                                            unsigned tile_d0, unsigned tile_d1,
                                            unsigned stride_d0) {
  const unsigned long long ga = (unsigned long long)gaddr;
  v4u g0;
  g0[0] = 1u;                                              // count=1 (valid user D#)
  g0[1] = lds_addr;                                        // LDS byte address
  g0[2] = (unsigned)(ga & 0xFFFFFFFFu);                    // global_addr[31:0]
  g0[3] = (unsigned)((ga >> 32) & 0x1FFFFFFu) | (2u << 30); // global_addr[56:32] | type=2
  v8i g1;
  g1[0] = (int)(0x00010000u        // data_size = 1 (2 bytes)
              | 0x00100000u        // pad_enable
              | (4u << 22)         // pad_interval: 32 DWORDs (128B)
              | (3u << 25));       // pad_amount:   4 DWORDs (16B)
  g1[1] = (int)((tensor_d0 & 0xFFFFu) << 16);                       // tensor_dim0[15:0]
  g1[2] = (int)((tensor_d0 >> 16) | ((tensor_d1 & 0xFFFFu) << 16)); // td0 hi | td1 lo
  g1[3] = (int)((tensor_d1 >> 16) | (tile_d0 << 16));               // td1 hi | tile_dim0
  g1[4] = (int)(tile_d1 & 0xFFFFu);                                 // tile_dim1 | tile_dim2=0
  g1[5] = (int)stride_d0;                                           // dim0 stride [31:0]
  g1[6] = 0;                                                        // stride hi | dim1 stride lo
  g1[7] = 0;
  v4i z4 = {0, 0, 0, 0};
#if defined(__clang_major__) && (__clang_major__ >= 23)
  v8i z8 = {0, 0, 0, 0, 0, 0, 0, 0};
  __builtin_amdgcn_tensor_load_to_lds(g0, g1, z4, z4, z8, 0);
#else
  __builtin_amdgcn_tensor_load_to_lds(g0, g1, z4, z4, 0);
#endif
}
#endif  // HAVE_TDM

// -------------------------- conversion kernels -----------------------------

__global__ void k_cvt(const float* __restrict__ in, unsigned short* __restrict__ out, int n) {
  int i = blockIdx.x * blockDim.x + threadIdx.x;
  if (i < n) out[i] = f32_to_bf16(in[i]);
}

// in: f32 [R][C] row-major  ->  out: bf16 [C][R]  (K-major weights / txt^T)
__global__ void k_cvt_t(const float* __restrict__ in, unsigned short* __restrict__ out,
                        int R, int C) {
  int i = blockIdx.x * blockDim.x + threadIdx.x;
  if (i < R * C) {
    int r = i / C, c = i - r * C;
    out[(size_t)c * R + r] = f32_to_bf16(in[i]);
  }
}

// ------------------------------ WMMA GEMM ----------------------------------
// C[M][N] = act(A[M][K] @ Wt[N][K]^T + bias), bf16 in, f32 accum, bf16 out.
// Block: 256 thr = 8 waves; block tile 64(M) x 128(N); wave tile 16 x 64.
// A tile (64xKT) via per-thread async-to-LDS; B tile (128xKT) via one TDM
// descriptor issued by wave 0. Both double-buffered.

__global__ __launch_bounds__(256) void k_gemm(
    const unsigned short* __restrict__ A,   // [M][K] bf16
    const unsigned short* __restrict__ Wt,  // [N][K] bf16 (pre-transposed weight)
    const float* __restrict__ bias,         // [N] f32
    unsigned short* __restrict__ C,         // [M][N] bf16
    int M, int N, int K, int relu) {
  extern __shared__ char smem[];
  unsigned short* Abuf = (unsigned short*)smem;        // [2][64][LDSTR]
  unsigned short* Bbuf = Abuf + 2 * A_ELE;             // [2][128][LDSTR]

  const int tid  = threadIdx.x;
  const int wave = tid >> 5;
  const int lane = tid & 31;
  const int wm   = wave >> 1;        // 0..3 -> 4 x 16 = 64 rows
  const int wn   = wave & 1;         // 0..1 -> 2 x 64 = 128 cols
  const int m0   = blockIdx.y * 64  + wm * 16;
  const int n0   = blockIdx.x * 128 + wn * 64;
  const int lx   = lane & 15;
  const int hi   = lane >> 4;
  const int kloA = hi * 16;
  const int kloB = hi * 16;

  const int rowA0 = blockIdx.y * 64;
  const int rowB0 = blockIdx.x * 128;

  const unsigned ldsA = lds_off_of(Abuf);
  const unsigned ldsB = lds_off_of(Bbuf);

  v8f acc0 = {}, acc1 = {}, acc2 = {}, acc3 = {};

  auto issue_A = [&](int k0, int b) {   // 64x64 bf16 = 512 x 16B chunks
    const unsigned la = ldsA + (unsigned)(b * A_ELE) * 2u;
#pragma unroll
    for (int c = 0; c < 2; ++c) {
      const int ch  = tid + c * 256;
      const int row = ch >> 3, cc = (ch & 7) * 8;
      async_ld_b128(la + (unsigned)(row * LDSTR + cc) * 2u,
                    A + (size_t)(rowA0 + row) * K + k0 + cc);
    }
  };

  auto issue_B = [&](int k0, int b) {
    const unsigned lb = ldsB + (unsigned)(b * B_ELE) * 2u;
#if HAVE_TDM
    if (wave == 0)
      tdm_load_2d(lb, Wt + (size_t)rowB0 * K + k0,
                  (unsigned)K, 128u, (unsigned)KT, 128u, (unsigned)K);
#else
#pragma unroll
    for (int c = 0; c < 4; ++c) {       // 128x64 bf16 = 1024 x 16B chunks
      const int ch  = tid + c * 256;
      const int row = ch >> 3, cc = (ch & 7) * 8;
      async_ld_b128(lb + (unsigned)(row * LDSTR + cc) * 2u,
                    Wt + (size_t)(rowB0 + row) * K + k0 + cc);
    }
#endif
  };

  // All 10 fragments of a K-tile loaded into distinct regs, then 8 WMMAs:
  // lets the compiler clause the ds_load_b128s and use partial DScnt waits.
  auto compute = [&](int b) {
    const unsigned short* At  = Abuf + b * A_ELE;
    const unsigned short* Bt  = Bbuf + b * B_ELE;
    const unsigned short* Ar  = At + (wm * 16 + lx) * LDSTR + kloA;
    const unsigned short* Br0 = Bt + (wn * 64 +  0 + lx) * LDSTR + kloB;
    const unsigned short* Br1 = Bt + (wn * 64 + 16 + lx) * LDSTR + kloB;
    const unsigned short* Br2 = Bt + (wn * 64 + 32 + lx) * LDSTR + kloB;
    const unsigned short* Br3 = Bt + (wn * 64 + 48 + lx) * LDSTR + kloB;

    V16BF a0, a1, b0, b1, b2, b3, b4, b5, b6, b7;
    a0.q[0] = *(const uint4*)(Ar +  0); a0.q[1] = *(const uint4*)(Ar +  8);
    a1.q[0] = *(const uint4*)(Ar + 32); a1.q[1] = *(const uint4*)(Ar + 40);
    b0.q[0] = *(const uint4*)(Br0 +  0); b0.q[1] = *(const uint4*)(Br0 +  8);
    b1.q[0] = *(const uint4*)(Br1 +  0); b1.q[1] = *(const uint4*)(Br1 +  8);
    b2.q[0] = *(const uint4*)(Br2 +  0); b2.q[1] = *(const uint4*)(Br2 +  8);
    b3.q[0] = *(const uint4*)(Br3 +  0); b3.q[1] = *(const uint4*)(Br3 +  8);
    b4.q[0] = *(const uint4*)(Br0 + 32); b4.q[1] = *(const uint4*)(Br0 + 40);
    b5.q[0] = *(const uint4*)(Br1 + 32); b5.q[1] = *(const uint4*)(Br1 + 40);
    b6.q[0] = *(const uint4*)(Br2 + 32); b6.q[1] = *(const uint4*)(Br2 + 40);
    b7.q[0] = *(const uint4*)(Br3 + 32); b7.q[1] = *(const uint4*)(Br3 + 40);

    acc0 = __builtin_amdgcn_wmma_f32_16x16x32_bf16(false, a0.v, false, b0.v, (short)0, acc0, false, false);
    acc1 = __builtin_amdgcn_wmma_f32_16x16x32_bf16(false, a0.v, false, b1.v, (short)0, acc1, false, false);
    acc2 = __builtin_amdgcn_wmma_f32_16x16x32_bf16(false, a0.v, false, b2.v, (short)0, acc2, false, false);
    acc3 = __builtin_amdgcn_wmma_f32_16x16x32_bf16(false, a0.v, false, b3.v, (short)0, acc3, false, false);
    acc0 = __builtin_amdgcn_wmma_f32_16x16x32_bf16(false, a1.v, false, b4.v, (short)0, acc0, false, false);
    acc1 = __builtin_amdgcn_wmma_f32_16x16x32_bf16(false, a1.v, false, b5.v, (short)0, acc1, false, false);
    acc2 = __builtin_amdgcn_wmma_f32_16x16x32_bf16(false, a1.v, false, b6.v, (short)0, acc2, false, false);
    acc3 = __builtin_amdgcn_wmma_f32_16x16x32_bf16(false, a1.v, false, b7.v, (short)0, acc3, false, false);
  };

  const int NTILES = K / KT;
  issue_A(0, 0);
  issue_B(0, 0);
  int buf = 0;

  for (int kt = 0; kt < NTILES - 1; ++kt) {   // steady state: next tile in flight
    issue_A((kt + 1) * KT, buf ^ 1);
    issue_B((kt + 1) * KT, buf ^ 1);
#if HAVE_TDM
    wait_asynccnt<2>();
    if (wave == 0) wait_tensorcnt<1>();
#else
    wait_asynccnt<6>();
#endif
    __syncthreads();
    compute(buf);
    __syncthreads();
    buf ^= 1;
  }
  wait_asynccnt<0>();
#if HAVE_TDM
  if (wave == 0) wait_tensorcnt<0>();
#endif
  __syncthreads();
  compute(buf);

  // Epilogue: C layout -> elem r: row = m0 + hi*8 + r, col = n0 + j*16 + lx
  v8f accs[4] = {acc0, acc1, acc2, acc3};
#pragma unroll
  for (int j = 0; j < 4; ++j) {
    const int ncol = n0 + j * 16 + lx;
    const float bv = bias[ncol];
#pragma unroll
    for (int r = 0; r < 8; ++r) {
      const int row = m0 + hi * 8 + r;
      float v = accs[j][r] + bv;
      if (relu) v = fmaxf(v, 0.0f);
      C[(size_t)row * N + ncol] = f32_to_bf16(v);
    }
  }
}

// ------------------------- fused CLIP loss kernel --------------------------
// Per block: 16 rows of p_img. WMMA 16x4096 sim tile into dynamic LDS (256KB),
// then row-wise sumsq / max / argmax / logsumexp. exp(logit_scale) cancels
// under the first row-normalize; second normalize of a unit row is a no-op.

__global__ __launch_bounds__(256) void k_clip(
    const unsigned short* __restrict__ Pimg,  // [B][D] bf16
    const unsigned short* __restrict__ Ptxt,  // [NTXT][D] bf16
    const int*   __restrict__ target,
    const float* __restrict__ temp_p,
    float* __restrict__ accum) {              // [0]=loss sum, [1]=acc count
  extern __shared__ char smem[];
  float*          SIM = (float*)smem;                                     // [16][N_TXT]
  unsigned short* AS  = (unsigned short*)(smem + (size_t)16 * N_TXT * 4); // [16][D_SZ]

  const int tid  = threadIdx.x;
  const int wave = tid >> 5;
  const int lane = tid & 31;
  const int r0   = blockIdx.x * 16;
  const int lx   = lane & 15;
  const int hi   = lane >> 4;

  // Async-stage the 16 p_img rows (16KB contiguous) into LDS.
  {
    const unsigned asBase = lds_off_of(AS);
    const unsigned short* src = Pimg + (size_t)r0 * D_SZ;
#pragma unroll
    for (int c = 0; c < 4; ++c) {             // 1024 chunks of 16B
      const int ch = tid + c * 256;
      async_ld_b128(asBase + (unsigned)ch * 16u, src + ch * 8);
    }
    wait_asynccnt<0>();
  }
  __syncthreads();

  // Phase 1: WMMA sim tile. Each wave covers 32 of the 256 column tiles.
  const unsigned short* Alds = AS + lx * D_SZ + hi * 16;
  for (int t = wave; t < N_TXT / 16; t += 8) {
    const int n0 = t * 16;
    v8f acc = {};
    const unsigned short* Brow = Ptxt + (size_t)(n0 + lx) * D_SZ + hi * 16;
#pragma unroll 2
    for (int k = 0; k < D_SZ; k += 64) {
      V16BF a0, a1, b0, b1;
      a0.q[0] = *(const uint4*)(Alds + k);      a0.q[1] = *(const uint4*)(Alds + k +  8);
      a1.q[0] = *(const uint4*)(Alds + k + 32); a1.q[1] = *(const uint4*)(Alds + k + 40);
      b0.q[0] = *(const uint4*)(Brow + k);      b0.q[1] = *(const uint4*)(Brow + k +  8);
      b1.q[0] = *(const uint4*)(Brow + k + 32); b1.q[1] = *(const uint4*)(Brow + k + 40);
      acc = __builtin_amdgcn_wmma_f32_16x16x32_bf16(false, a0.v, false, b0.v, (short)0, acc, false, false);
      acc = __builtin_amdgcn_wmma_f32_16x16x32_bf16(false, a1.v, false, b1.v, (short)0, acc, false, false);
    }
#pragma unroll
    for (int r = 0; r < 8; ++r)
      SIM[(size_t)(hi * 8 + r) * N_TXT + n0 + lx] = acc[r];
  }
  __syncthreads();

  // Phase 2: row reductions. Wave w handles rows 2w, 2w+1.
  float lossP = 0.0f, accP = 0.0f;
  const float temp = temp_p[0];
#pragma unroll
  for (int rr = 0; rr < 2; ++rr) {
    const int row = wave * 2 + rr;
    const float* srow = SIM + (size_t)row * N_TXT;
    float ss = 0.0f, mx = -3.4e38f;
    int am = 0;
    for (int c = lane; c < N_TXT; c += 32) {
      float v = srow[c];
      ss += v * v;
      if (v > mx || (v == mx && c < am)) { mx = v; am = c; }
    }
    for (int off = 16; off; off >>= 1) {
      ss += __shfl_xor(ss, off, 32);
      float omx = __shfl_xor(mx, off, 32);
      int   oam = __shfl_xor(am, off, 32);
      if (omx > mx || (omx == mx && oam < am)) { mx = omx; am = oam; }
    }
    const float norm    = sqrtf(ss);
    const float scale   = 1.0f / (norm * temp);
    const float mscaled = mx * scale;
    float se = 0.0f;
    for (int c = lane; c < N_TXT; c += 32)
      se += __expf(srow[c] * scale - mscaled);
    for (int off = 16; off; off >>= 1) se += __shfl_xor(se, off, 32);
    if (lane == 0) {
      const int   tgt  = target[r0 + row];
      const float tv   = srow[tgt];
      const float logZ = mscaled + __logf(se);
      lossP += -(tv * scale - logZ);
      accP  += (am == tgt) ? 1.0f : 0.0f;
    }
  }
  if (lane == 0) {
    atomicAdd(&accum[0], lossP);
    atomicAdd(&accum[1], accP);
  }
}

// ------------------------------ init / final -------------------------------

__global__ void k_init(float* accum) {
  if (threadIdx.x < 2) accum[threadIdx.x] = 0.0f;
}

__global__ void k_final(const float* __restrict__ accum, float* __restrict__ out) {
  if (threadIdx.x == 0) {
    out[0] = accum[0] / (float)B_SZ;  // loss
    out[1] = accum[1];                // acc
  }
}

// ------------------------------ launch -------------------------------------

extern "C" void kernel_launch(void* const* d_in, const int* in_sizes, int n_in,
                              void* d_out, int out_size, void* d_ws, size_t ws_size,
                              hipStream_t stream) {
  const float* img  = (const float*)d_in[0];
  const float* txt  = (const float*)d_in[1];
  const int*   tgt  = (const int*)  d_in[2];
  const float* temp = (const float*)d_in[3];
  const float* Wi0 = (const float*)d_in[4];  const float* bi0 = (const float*)d_in[5];
  const float* Wi1 = (const float*)d_in[6];  const float* bi1 = (const float*)d_in[7];
  const float* Wi2 = (const float*)d_in[8];  const float* bi2 = (const float*)d_in[9];
  const float* Wt0 = (const float*)d_in[10]; const float* bt0 = (const float*)d_in[11];
  const float* Wt1 = (const float*)d_in[12]; const float* bt1 = (const float*)d_in[13];
  const float* Wt2 = (const float*)d_in[14]; const float* bt2 = (const float*)d_in[15];
  (void)in_sizes; (void)n_in; (void)out_size; (void)ws_size;

  char* ws = (char*)d_ws;
  size_t off = 0;
  auto alloc = [&](size_t bytes) -> char* {
    char* p = ws + off;
    off += (bytes + 255) & ~(size_t)255;
    return p;
  };

  unsigned short* Xi   = (unsigned short*)alloc((size_t)B_SZ  * D_SZ * 2);
  unsigned short* Xt   = (unsigned short*)alloc((size_t)N_TXT * D_SZ * 2);
  unsigned short* Wi0t = (unsigned short*)alloc((size_t)D_SZ * H_SZ * 2);
  unsigned short* Wi1t = (unsigned short*)alloc((size_t)H_SZ * H_SZ * 2);
  unsigned short* Wi2t = (unsigned short*)alloc((size_t)H_SZ * D_SZ * 2);
  unsigned short* Wt0t = (unsigned short*)alloc((size_t)D_SZ * H_SZ * 2);
  unsigned short* Wt1t = (unsigned short*)alloc((size_t)H_SZ * H_SZ * 2);
  unsigned short* Wt2t = (unsigned short*)alloc((size_t)H_SZ * D_SZ * 2);
  unsigned short* H1   = (unsigned short*)alloc((size_t)B_SZ  * H_SZ * 2);
  unsigned short* H2   = (unsigned short*)alloc((size_t)B_SZ  * H_SZ * 2);
  unsigned short* Pi   = (unsigned short*)alloc((size_t)B_SZ  * D_SZ * 2);
  unsigned short* T1   = (unsigned short*)alloc((size_t)N_TXT * H_SZ * 2);
  unsigned short* T2   = (unsigned short*)alloc((size_t)N_TXT * H_SZ * 2);
  unsigned short* Pt   = (unsigned short*)alloc((size_t)N_TXT * D_SZ * 2);
  float*          accum = (float*)alloc(256);

  // bf16 conversions (weights K-major transposed so WMMA B-frags are contiguous)
  k_cvt  <<<(B_SZ * D_SZ + 255) / 256, 256, 0, stream>>>(img, Xi, B_SZ * D_SZ);
  k_cvt_t<<<(D_SZ * N_TXT + 255) / 256, 256, 0, stream>>>(txt, Xt, D_SZ, N_TXT);
  k_cvt_t<<<(D_SZ * H_SZ + 255) / 256, 256, 0, stream>>>(Wi0, Wi0t, D_SZ, H_SZ);
  k_cvt_t<<<(H_SZ * H_SZ + 255) / 256, 256, 0, stream>>>(Wi1, Wi1t, H_SZ, H_SZ);
  k_cvt_t<<<(H_SZ * D_SZ + 255) / 256, 256, 0, stream>>>(Wi2, Wi2t, H_SZ, D_SZ);
  k_cvt_t<<<(D_SZ * H_SZ + 255) / 256, 256, 0, stream>>>(Wt0, Wt0t, D_SZ, H_SZ);
  k_cvt_t<<<(H_SZ * H_SZ + 255) / 256, 256, 0, stream>>>(Wt1, Wt1t, H_SZ, H_SZ);
  k_cvt_t<<<(H_SZ * D_SZ + 255) / 256, 256, 0, stream>>>(Wt2, Wt2t, H_SZ, D_SZ);

  const size_t gsh = (size_t)(2 * A_ELE + 2 * B_ELE) * 2;  // 54KB double-buffered tiles
  // img MLP
  k_gemm<<<dim3(H_SZ / 128, B_SZ / 64), 256, gsh, stream>>>(Xi, Wi0t, bi0, H1, B_SZ, H_SZ, D_SZ, 1);
  k_gemm<<<dim3(H_SZ / 128, B_SZ / 64), 256, gsh, stream>>>(H1, Wi1t, bi1, H2, B_SZ, H_SZ, H_SZ, 1);
  k_gemm<<<dim3(D_SZ / 128, B_SZ / 64), 256, gsh, stream>>>(H2, Wi2t, bi2, Pi, B_SZ, D_SZ, H_SZ, 0);
  // txt MLP
  k_gemm<<<dim3(H_SZ / 128, N_TXT / 64), 256, gsh, stream>>>(Xt, Wt0t, bt0, T1, N_TXT, H_SZ, D_SZ, 1);
  k_gemm<<<dim3(H_SZ / 128, N_TXT / 64), 256, gsh, stream>>>(T1, Wt1t, bt1, T2, N_TXT, H_SZ, H_SZ, 1);
  k_gemm<<<dim3(D_SZ / 128, N_TXT / 64), 256, gsh, stream>>>(T2, Wt2t, bt2, Pt, N_TXT, D_SZ, H_SZ, 0);

  // fused sim + double-normalize + log-softmax + argmax
  k_init<<<1, 32, 0, stream>>>(accum);
  const size_t csh = (size_t)16 * N_TXT * 4 + (size_t)16 * D_SZ * 2;  // 272KB < 320KB WGP LDS
  k_clip<<<B_SZ / 16, 256, csh, stream>>>(Pi, Pt, tgt, temp, accum);
  k_final<<<1, 32, 0, stream>>>(accum, (float*)d_out);
}
